// Net_79585743995370
// MI455X (gfx1250) — matrix operand order repaired and checked by reading, not compiled, not run on hardware
//
#include <hip/hip_runtime.h>

// 2-layer GAT pipeline for MI455X (gfx1250, wave32, WMMA).
// e_proj is only consumed via (e_full @ a_e), so both E x 4000 x 32 GEMMs
// collapse to GEMVs against (We @ a_e); one fused pass over edge_attr
// (640 MB -> ~27.5 us HBM floor) computes both layers' edge scalars with
// V_WMMA_F32_16X16X32_BF16 (fp32 accumulate).

#define EDK 4000   // edge feature dim
#define CCH 32     // channels
#define FIN 128    // input node feature dim
#define NEG_SLOPE 0.2f

typedef __attribute__((ext_vector_type(16))) __bf16 v16bf;
typedef __attribute__((ext_vector_type(8)))  float  v8f;

static __device__ __forceinline__ void atomicMaxFloat(float* a, float v) {
    // monotonic-bits trick: valid for all non-NaN floats, init must be -large
    if (v >= 0.0f) atomicMax((int*)a, __float_as_int(v));
    else           atomicMin((unsigned int*)a, __float_as_uint(v));
}

__global__ void k_fill(float* __restrict__ p, float v, int n) {
    int t = blockIdx.x * blockDim.x + threadIdx.x;
    if (t < n) p[t] = v;
}

// w1v[k] = sum_c We1[k,c]*a_e1[c] ; w2v likewise   (2*4000 threads)
__global__ void k_gemv_w(const float* __restrict__ We1, const float* __restrict__ ae1,
                         const float* __restrict__ We2, const float* __restrict__ ae2,
                         float* __restrict__ w1v, float* __restrict__ w2v) {
    int t = blockIdx.x * blockDim.x + threadIdx.x;
    if (t >= 2 * EDK) return;
    const float* We = (t < EDK) ? We1 : We2;
    const float* ae = (t < EDK) ? ae1 : ae2;
    float*       o  = (t < EDK) ? w1v : w2v;
    int k = (t < EDK) ? t : t - EDK;
    float s = 0.0f;
#pragma unroll
    for (int c = 0; c < CCH; ++c) s = fmaf(We[k * CCH + c], ae[c], s);
    o[k] = s;
}

// Build A fragment (16x32 bf16, MxK layout) from 16 contiguous fp32 loads:
// lane holds K = k0 + 16g + {0..15} (elems 0-15).
static __device__ __forceinline__ v16bf load_a_frag(const float* rp, int k0, int g) {
    const float4 f0 = *(const float4*)(rp + k0 + g * 16);
    const float4 f1 = *(const float4*)(rp + k0 + g * 16 + 4);
    const float4 f2 = *(const float4*)(rp + k0 + g * 16 + 8);
    const float4 f3 = *(const float4*)(rp + k0 + g * 16 + 12);
    v16bf a;
    a[0]  = (__bf16)f0.x; a[1]  = (__bf16)f0.y; a[2]  = (__bf16)f0.z; a[3]  = (__bf16)f0.w;
    a[4]  = (__bf16)f1.x; a[5]  = (__bf16)f1.y; a[6]  = (__bf16)f1.z; a[7]  = (__bf16)f1.w;
    a[8]  = (__bf16)f2.x; a[9]  = (__bf16)f2.y; a[10] = (__bf16)f2.z; a[11] = (__bf16)f2.w;
    a[12] = (__bf16)f3.x; a[13] = (__bf16)f3.y; a[14] = (__bf16)f3.z; a[15] = (__bf16)f3.w;
    return a;
}

// ev1[i] = edge_attr[i,:] . w1v ; ev2[i] = edge_attr[i,:] . w2v
// One wave per 32-edge tile (two 16-row WMMA tiles sharing each B fragment).
// B matrix: col 0 = w1v chunk, col 1 = w2v chunk; cols 2..15 carry garbage
// (lanes n>=2 load from lw1 too) -- harmless, those D columns are discarded.
__global__ __launch_bounds__(256) void
k_edge_dot(const float* __restrict__ edge_attr,
           const float* __restrict__ w1v, const float* __restrict__ w2v,
           float* __restrict__ ev1, float* __restrict__ ev2, int E) {
    __shared__ __bf16 lw1[EDK];
    __shared__ __bf16 lw2[EDK];
    for (int k = threadIdx.x; k < EDK; k += blockDim.x) {
        lw1[k] = (__bf16)w1v[k];
        lw2[k] = (__bf16)w2v[k];
    }
    __syncthreads();

    const int wave = threadIdx.x >> 5;
    const int lane = threadIdx.x & 31;
    const int g    = lane >> 4;   // lane group (hi/lo half of wave)
    const int nco  = lane & 15;   // B/D column, also A row within tile
    const long tile = blockIdx.x * (blockDim.x >> 5) + wave;   // 32-edge tile

    long row0 = tile * 32 + nco;
    long row1 = tile * 32 + 16 + nco;
    if (row0 >= E) row0 = E - 1;                 // clamp; stores guarded below
    if (row1 >= E) row1 = E - 1;
    const float* rp0 = edge_attr + row0 * (long)EDK;
    const float* rp1 = edge_attr + row1 * (long)EDK;

    // lanes 0/1 supply the real columns; others fetch don't-care data (no branch)
    const __bf16* bp = (nco == 1) ? lw2 : lw1;

    v8f acc0 = {};
    v8f acc1 = {};
    for (int k0 = 0; k0 < EDK; k0 += 32) {
        __builtin_prefetch(rp0 + k0 + 256, 0, 0);   // speculative; OOB is dropped
        __builtin_prefetch(rp1 + k0 + 256, 0, 0);
        const v16bf a0 = load_a_frag(rp0, k0, g);
        const v16bf a1 = load_a_frag(rp1, k0, g);
        // B fragment (32x16 bf16): lane n holds K = k0 + 16g + {0..15} of col n
        const v16bf b = *(const v16bf*)(bp + k0 + g * 16);
        acc0 = __builtin_amdgcn_wmma_f32_16x16x32_bf16(false, a0, false, b,
                                                       (short)0, acc0, false, false);
        acc1 = __builtin_amdgcn_wmma_f32_16x16x32_bf16(false, a1, false, b,
                                                       (short)0, acc1, false, false);
    }
    // D layout: lane (g,n), VGPR r -> M = r + 8g, N = n. Only cols 0,1 useful.
    if (nco < 2) {
        float* o = (nco == 0) ? ev1 : ev2;
#pragma unroll
        for (int r = 0; r < 8; ++r) {
            long e0 = tile * 32 + r + 8 * g;
            long e1 = tile * 32 + 16 + r + 8 * g;
            if (e0 < E) o[e0] = acc0[r];
            if (e1 < E) o[e1] = acc1[r];
        }
    }
}

// H[n,c] = sum_k X[n,k] * W[k,c]     (one thread per output element)
template <int CIN>
__global__ void k_node_mm(const float* __restrict__ X, const float* __restrict__ W,
                          float* __restrict__ H, int N) {
    int t = blockIdx.x * blockDim.x + threadIdx.x;
    if (t >= N * CCH) return;
    int n = t >> 5, c = t & 31;
    const float* xr = X + (long)n * CIN;
    float s = 0.0f;
#pragma unroll 8
    for (int k = 0; k < CIN; ++k) s = fmaf(xr[k], W[k * CCH + c], s);
    H[t] = s;
}

__global__ void k_hsd(const float* __restrict__ H, const float* __restrict__ asrc,
                      const float* __restrict__ adst, float* __restrict__ hs,
                      float* __restrict__ hd, int N) {
    int n = blockIdx.x * blockDim.x + threadIdx.x;
    if (n >= N) return;
    const float* hr = H + (long)n * CCH;
    float s = 0.0f, d = 0.0f;
#pragma unroll
    for (int c = 0; c < CCH; ++c) { s = fmaf(hr[c], asrc[c], s); d = fmaf(hr[c], adst[c], d); }
    hs[n] = s; hd[n] = d;
}

// per-dst counts + sums of edge scalars (shared by both layers)
__global__ void k_cnt(const int* __restrict__ dst, const float* __restrict__ ev1,
                      const float* __restrict__ ev2, float* __restrict__ cnt,
                      float* __restrict__ s1, float* __restrict__ s2, int E) {
    int i = blockIdx.x * blockDim.x + threadIdx.x;
    if (i >= E) return;
    int d = dst[i];
    atomicAdd(&cnt[d], 1.0f);
    atomicAdd(&s1[d], ev1[i]);
    atomicAdd(&s2[d], ev2[i]);
}

__global__ void k_mean(const float* __restrict__ s1, const float* __restrict__ s2,
                       const float* __restrict__ cnt, float* __restrict__ lev1,
                       float* __restrict__ lev2, int N) {
    int n = blockIdx.x * blockDim.x + threadIdx.x;
    if (n >= N) return;
    float c = fmaxf(cnt[n], 1.0f);
    lev1[n] = s1[n] / c;
    lev2[n] = s2[n] / c;
}

// logits + leaky relu + segment max (over E real edges + N self loops)
__global__ void k_alpha(const int* __restrict__ src, const int* __restrict__ dst,
                        const float* __restrict__ hs, const float* __restrict__ hd,
                        const float* __restrict__ ev, const float* __restrict__ lev,
                        float* __restrict__ alpha, float* __restrict__ m, int E, int N) {
    int i = blockIdx.x * blockDim.x + threadIdx.x;
    if (i >= E + N) return;
    int s, d; float e;
    if (i < E) { s = src[i]; d = dst[i]; e = ev[i]; }
    else       { s = d = i - E;          e = lev[i - E]; }
    float a = hs[s] + hd[d] + e;
    a = (a > 0.0f) ? a : NEG_SLOPE * a;
    alpha[i] = a;
    atomicMaxFloat(&m[d], a);
}

__global__ void k_exp(const int* __restrict__ dst, float* __restrict__ alpha,
                      const float* __restrict__ m, float* __restrict__ denom,
                      int E, int N) {
    int i = blockIdx.x * blockDim.x + threadIdx.x;
    if (i >= E + N) return;
    int d = (i < E) ? dst[i] : (i - E);
    float a = __expf(alpha[i] - m[d]);
    alpha[i] = a;
    atomicAdd(&denom[d], a);
}

__global__ void k_bias_init(float* __restrict__ out, const float* __restrict__ b, int N) {
    int t = blockIdx.x * blockDim.x + threadIdx.x;
    if (t >= N * CCH) return;
    out[t] = b[t & 31];
}

// out[dst,c] += h[src,c] * alpha / (denom[dst]+eps)  (thread per (edge, channel))
__global__ void k_agg(const int* __restrict__ src, const int* __restrict__ dst,
                      const float* __restrict__ alpha, const float* __restrict__ denom,
                      const float* __restrict__ H, float* __restrict__ out, int E, int N) {
    int t = blockIdx.x * blockDim.x + threadIdx.x;
    if (t >= (E + N) * CCH) return;
    int i = t >> 5, c = t & 31;
    int s, d;
    if (i < E) { s = src[i]; d = dst[i]; }
    else       { s = d = i - E; }
    float w = alpha[i] / (denom[d] + 1e-16f);
    atomicAdd(&out[(long)d * CCH + c], H[(long)s * CCH + c] * w);
}

__global__ void k_relu(float* __restrict__ p, int n) {
    int t = blockIdx.x * blockDim.x + threadIdx.x;
    if (t < n) p[t] = fmaxf(p[t], 0.0f);
}

static inline int cdiv(long a, long b) { return (int)((a + b - 1) / b); }

extern "C" void kernel_launch(void* const* d_in, const int* in_sizes, int n_in,
                              void* d_out, int out_size, void* d_ws, size_t ws_size,
                              hipStream_t stream) {
    const float* x    = (const float*)d_in[0];
    const int*   eidx = (const int*)d_in[1];
    const float* eatt = (const float*)d_in[2];
    const float* W1   = (const float*)d_in[3];
    const float* as1  = (const float*)d_in[4];
    const float* ad1  = (const float*)d_in[5];
    const float* We1  = (const float*)d_in[6];
    const float* ae1  = (const float*)d_in[7];
    const float* b1   = (const float*)d_in[8];
    const float* W2   = (const float*)d_in[9];
    const float* as2  = (const float*)d_in[10];
    const float* ad2  = (const float*)d_in[11];
    const float* We2  = (const float*)d_in[12];
    const float* ae2  = (const float*)d_in[13];
    const float* b2   = (const float*)d_in[14];

    const int N = in_sizes[0] / FIN;
    const int E = in_sizes[1] / 2;
    const int* src = eidx;
    const int* dst = eidx + E;

    // workspace partition
    float* ws   = (float*)d_ws;
    float* w1v  = ws;  ws += EDK;
    float* w2v  = ws;  ws += EDK;
    float* ev1  = ws;  ws += E;
    float* ev2  = ws;  ws += E;
    float* h    = ws;  ws += (long)N * CCH;
    float* z    = ws;  ws += (long)N * CCH;
    float* hs   = ws;  ws += N;
    float* hd   = ws;  ws += N;
    float* cnt  = ws;  ws += N;
    float* s1   = ws;  ws += N;
    float* s2   = ws;  ws += N;
    float* lev1 = ws;  ws += N;
    float* lev2 = ws;  ws += N;
    float* alpha= ws;  ws += (E + N);
    float* mx   = ws;  ws += N;
    float* den  = ws;  ws += N;
    float* out  = (float*)d_out;

    const int B = 256;

    // ---- shared precompute (both layers) ----
    k_gemv_w<<<cdiv(2 * EDK, B), B, 0, stream>>>(We1, ae1, We2, ae2, w1v, w2v);

    {   // fused edge-scalar pass: the only big-memory kernel (640 MB)
        int tiles  = cdiv(E, 32);             // 32-edge tiles, one per wave
        int blocks = cdiv(tiles, B / 32);
        k_edge_dot<<<blocks, B, 0, stream>>>(eatt, w1v, w2v, ev1, ev2, E);
    }

    k_fill<<<cdiv(N, B), B, 0, stream>>>(cnt, 0.0f, N);
    k_fill<<<cdiv(N, B), B, 0, stream>>>(s1, 0.0f, N);
    k_fill<<<cdiv(N, B), B, 0, stream>>>(s2, 0.0f, N);
    k_cnt<<<cdiv(E, B), B, 0, stream>>>(dst, ev1, ev2, cnt, s1, s2, E);
    k_mean<<<cdiv(N, B), B, 0, stream>>>(s1, s2, cnt, lev1, lev2, N);

    // ---- layer 1 ----
    k_node_mm<FIN><<<cdiv((long)N * CCH, B), B, 0, stream>>>(x, W1, h, N);
    k_hsd<<<cdiv(N, B), B, 0, stream>>>(h, as1, ad1, hs, hd, N);
    k_fill<<<cdiv(N, B), B, 0, stream>>>(mx, -3.0e38f, N);
    k_fill<<<cdiv(N, B), B, 0, stream>>>(den, 0.0f, N);
    k_alpha<<<cdiv(E + N, B), B, 0, stream>>>(src, dst, hs, hd, ev1, lev1, alpha, mx, E, N);
    k_exp<<<cdiv(E + N, B), B, 0, stream>>>(dst, alpha, mx, den, E, N);
    k_bias_init<<<cdiv((long)N * CCH, B), B, 0, stream>>>(z, b1, N);
    k_agg<<<cdiv((long)(E + N) * CCH, B), B, 0, stream>>>(src, dst, alpha, den, h, z, E, N);
    k_relu<<<cdiv((long)N * CCH, B), B, 0, stream>>>(z, N * CCH);

    // ---- layer 2 ----
    k_node_mm<CCH><<<cdiv((long)N * CCH, B), B, 0, stream>>>(z, W2, h, N);
    k_hsd<<<cdiv(N, B), B, 0, stream>>>(h, as2, ad2, hs, hd, N);
    k_fill<<<cdiv(N, B), B, 0, stream>>>(mx, -3.0e38f, N);
    k_fill<<<cdiv(N, B), B, 0, stream>>>(den, 0.0f, N);
    k_alpha<<<cdiv(E + N, B), B, 0, stream>>>(src, dst, hs, hd, ev2, lev2, alpha, mx, E, N);
    k_exp<<<cdiv(E + N, B), B, 0, stream>>>(dst, alpha, mx, den, E, N);
    k_bias_init<<<cdiv((long)N * CCH, B), B, 0, stream>>>(out, b2, N);
    k_agg<<<cdiv((long)(E + N) * CCH, B), B, 0, stream>>>(src, dst, alpha, den, h, out, E, N);
}
